// CrossAttention_36515811950660
// MI455X (gfx1250) — matrix operand order repaired and checked by reading, not compiled, not run on hardware
//
#include <hip/hip_runtime.h>
#include <math.h>

// ---------------------------------------------------------------------------
// CDNA5 (gfx1250) cross-attention:
//   out = softmax((qWq')(kWk')^T / 32) (vWv') Wo' + bo
// B=4, N=M=2048, DIM=1024, H=16, hd=64.
// All matmuls: v_wmma_f32_16x16x32_bf16 (wave32).
// Shared operands staged in LDS via async global->LDS loads (ASYNCcnt),
// double-buffered against the WMMA loop.
// ---------------------------------------------------------------------------

typedef __attribute__((ext_vector_type(16))) __bf16 v16bf;
typedef __attribute__((ext_vector_type(8)))  float  v8f;
typedef int v4i_vs __attribute__((vector_size(16)));   // matches builtin proto

#define DIM   1024
#define HEADS 16
#define HD    64
#define SEQ   2048
#define BATCH 4
#define MROWS (BATCH * SEQ)          // 8192 flattened rows

// float -> bf16, round-to-nearest-even
__device__ __forceinline__ __bf16 f2bf(float f) {
  union { float f; unsigned u; } v; v.f = f;
  unsigned r = (v.u + 0x7FFFu + ((v.u >> 16) & 1u)) >> 16;
  unsigned short s = (unsigned short)r;
  __bf16 b;
  __builtin_memcpy(&b, &s, 2);
  return b;
}

// 16-byte global -> LDS copy, async if the toolchain exposes it
__device__ __forceinline__ void async_cp16(const __bf16* g, __bf16* l) {
#if __has_builtin(__builtin_amdgcn_global_load_async_to_lds_b128)
  __builtin_amdgcn_global_load_async_to_lds_b128(
      (__attribute__((address_space(1))) v4i_vs*)g,
      (__attribute__((address_space(3))) v4i_vs*)l, 0, 0);
#else
  uint4 t = *(const uint4*)g;
  *(uint4*)l = t;
#endif
}

__device__ __forceinline__ void wait_async0() {
#if __has_builtin(__builtin_amdgcn_s_wait_asynccnt)
  __builtin_amdgcn_s_wait_asynccnt(0);
#else
  asm volatile("s_wait_asynccnt 0" ::: "memory");
#endif
}

__device__ __forceinline__ void wait_ds0() {
  asm volatile("s_wait_dscnt 0" ::: "memory");
}

// ---- WMMA fragment loaders (ISA 7.12.2 layouts, wave32) -------------------
// A (16x32 bf16, row-major src, stride ld): lane&15 = M row;
// lanes 0-15: K = {0..7,16..23}; lanes 16-31: K = {8..15,24..31}.
__device__ __forceinline__ v16bf load_a_bf16(const __bf16* __restrict__ p,
                                             int ld, int row0, int k0) {
  int lane = threadIdx.x & 31;
  int half = lane >> 4, m = lane & 15;
  const __bf16* r = p + (size_t)(row0 + m) * ld;
  int kA = k0 + 8 * half;
  int kB = k0 + 16 + 8 * half;
  v16bf a;
#pragma unroll
  for (int e = 0; e < 8; ++e) { a[e] = r[kA + e]; a[8 + e] = r[kB + e]; }
  return a;
}

// B (32x16 bf16): B(k,n) = src[n][k], src row-major stride ld.
// lane&15 = N column; lanes 0-15: K = {0..7,16..23}; lanes 16-31: K = {8..15,24..31}.
__device__ __forceinline__ v16bf load_b_bf16(const __bf16* __restrict__ p,
                                             int ld, int n0, int k0) {
  int lane = threadIdx.x & 31;
  int half = lane >> 4, n = lane & 15;
  const __bf16* r = p + (size_t)(n0 + n) * ld + k0;
  v16bf b;
#pragma unroll
  for (int e = 0; e < 8; ++e) { b[e] = r[8 * half + e]; b[8 + e] = r[16 + 8 * half + e]; }
  return b;
}

#define WMMA_BF16(A, B, C) \
  __builtin_amdgcn_wmma_f32_16x16x32_bf16(false, (A), false, (B), (short)0, (C), false, false)

__device__ __forceinline__ float rowmax16(float v) {
#pragma unroll
  for (int m = 1; m < 16; m <<= 1) v = fmaxf(v, __shfl_xor(v, m, 32));
  return v;
}
__device__ __forceinline__ float rowsum16(float v) {
#pragma unroll
  for (int m = 1; m < 16; m <<= 1) v += __shfl_xor(v, m, 32);
  return v;
}

// ---------------------------------------------------------------------------
// Kernel 1: fp32 -> bf16 convert
__global__ void ca_cvt_bf16(const float* __restrict__ in, __bf16* __restrict__ out, int n) {
  int i = blockIdx.x * blockDim.x + threadIdx.x;
  if (i < n) out[i] = f2bf(in[i]);
}

// ---------------------------------------------------------------------------
// Kernel 2: projection GEMM  Y = X(8192x1024) * W^T + bias  (bf16 out).
// 8-wave block computes 128x64; the shared W panel is async double-buffered in LDS.
// mode 0: dst[b][h][seq][d]  (Q,K)    mode 1: dst[b][h][d][seq]  (V -> Vt)
__global__ void ca_proj(const __bf16* __restrict__ X, const __bf16* __restrict__ W,
                        const float* __restrict__ bias, __bf16* __restrict__ dst, int mode) {
  __shared__ __align__(16) __bf16 bbuf[2][64 * 32];   // [n 64][k 32]

  int tid  = threadIdx.x;
  int wave = tid >> 5;
  int lane = tid & 31;
  int half = lane >> 4, nl = lane & 15;
  int row0 = blockIdx.x * 128 + wave * 16;
  int col0 = blockIdx.y * 64;

  // staging map: each thread moves 16B; wrow = tid/4 in [0,64), wcol = (tid&3)*8
  const __bf16* wsrc = W + (size_t)(col0 + (tid >> 2)) * DIM + (tid & 3) * 8;
  __bf16* ldst0 = &bbuf[0][tid * 8];
  __bf16* ldst1 = &bbuf[1][tid * 8];

  v8f zero = {0.f, 0.f, 0.f, 0.f, 0.f, 0.f, 0.f, 0.f};
  v8f acc[4];
#pragma unroll
  for (int t = 0; t < 4; ++t) acc[t] = zero;

  async_cp16(wsrc, ldst0);                   // prefetch k=0
  wait_async0();
  __syncthreads();

  const int NK = DIM / 32;                   // 32 stages
  for (int ks = 0; ks < NK; ++ks) {
    int k = ks * 32;
    const __bf16* bcur = bbuf[ks & 1];
    if (ks + 1 < NK)                         // overlap next stage with compute
      async_cp16(wsrc + k + 32, (ks & 1) ? ldst0 : ldst1);

    v16bf a = load_a_bf16(X, DIM, row0, k);  // per-wave rows, direct from L2
#pragma unroll
    for (int t = 0; t < 4; ++t) {
      v16bf b = load_b_bf16(bcur, 32, 16 * t, 0);
      acc[t] = WMMA_BF16(a, b, acc[t]);
    }
    if (ks + 1 < NK) wait_async0();
    __syncthreads();
  }

#pragma unroll
  for (int t = 0; t < 4; ++t) {
    int col = col0 + 16 * t + nl;
    float bv = bias[col];
    int h = col >> 6, d = col & 63;
#pragma unroll
    for (int r = 0; r < 8; ++r) {
      int m = row0 + r + 8 * half;
      int batch = m >> 11, seq = m & (SEQ - 1);
      float val = acc[t][r] + bv;
      if (mode == 0)
        dst[(((size_t)batch * HEADS + h) * SEQ + seq) * HD + d] = f2bf(val);
      else
        dst[(((size_t)batch * HEADS + h) * HD + d) * SEQ + seq] = f2bf(val);
    }
  }
}

// ---------------------------------------------------------------------------
// Kernel 3: flash attention. 8-wave block = 128 query rows of one (b,h).
// K-block (32x64) and Vt-block (64x32) async double-buffered in LDS, shared
// by all 8 waves. P restaged per-wave through LDS (C->A layout).
__global__ void ca_attn(const __bf16* __restrict__ Qh, const __bf16* __restrict__ Kh,
                        const __bf16* __restrict__ Vt, __bf16* __restrict__ Xctx) {
  __shared__ __align__(16) __bf16 kbuf[2][32 * 64];   // [key 32][d 64]
  __shared__ __align__(16) __bf16 vbuf[2][64 * 32];   // [d 64][key 32]
  __shared__ __align__(16) __bf16 pbuf[8][16 * 32];   // per-wave P tile

  int bh = blockIdx.y;                  // b*16 + h
  int batch = bh >> 4, h = bh & 15;
  int tid  = threadIdx.x;
  int wave = tid >> 5;
  int lane = tid & 31;
  int half = lane >> 4, nl = lane & 15;
  int row0 = blockIdx.x * 128 + wave * 16;   // within SEQ

  const __bf16* Qp = Qh + (size_t)bh * SEQ * HD;
  const __bf16* Kp = Kh + (size_t)bh * SEQ * HD;
  const __bf16* Vp = Vt + (size_t)bh * HD * SEQ;      // [d][seq]

  // staging maps (16B per thread per buffer)
  const __bf16* ksrc = Kp + (size_t)(tid >> 3) * HD + (tid & 7) * 8;  // + key0*HD
  const __bf16* vsrc = Vp + (size_t)(tid >> 2) * SEQ + (tid & 3) * 8; // + key0
  __bf16* kdst[2] = { &kbuf[0][tid * 8], &kbuf[1][tid * 8] };
  __bf16* vdst[2] = { &vbuf[0][tid * 8], &vbuf[1][tid * 8] };

  v16bf aq0 = load_a_bf16(Qp, HD, row0, 0);
  v16bf aq1 = load_a_bf16(Qp, HD, row0, 32);

  v8f zero = {0.f, 0.f, 0.f, 0.f, 0.f, 0.f, 0.f, 0.f};
  v8f o[4];
#pragma unroll
  for (int t = 0; t < 4; ++t) o[t] = zero;
  float mrow[8], lrow[8];
#pragma unroll
  for (int r = 0; r < 8; ++r) { mrow[r] = -3.0e38f; lrow[r] = 0.f; }

  const float scale = 0.03125f;         // DIM^-0.5 = 1/32 (full dim, per reference)

  async_cp16(ksrc, kdst[0]);            // prefetch key-block 0
  async_cp16(vsrc, vdst[0]);
  wait_async0();
  __syncthreads();

  const int NS = SEQ / 32;              // 64 key blocks
  for (int s = 0; s < NS; ++s) {
    int key0 = s * 32, cur = s & 1;
    if (s + 1 < NS) {                   // overlap next K/V stage with compute
      async_cp16(ksrc + (size_t)(key0 + 32) * HD, kdst[cur ^ 1]);
      async_cp16(vsrc + key0 + 32,               vdst[cur ^ 1]);
    }

    v8f sacc[2];
#pragma unroll
    for (int j = 0; j < 2; ++j) {
      v16bf b0 = load_b_bf16(kbuf[cur], HD, 16 * j, 0);
      v16bf b1 = load_b_bf16(kbuf[cur], HD, 16 * j, 32);
      v8f sj = zero;
      sj = WMMA_BF16(aq0, b0, sj);
      sj = WMMA_BF16(aq1, b1, sj);
      sacc[j] = sj;
    }

    __bf16* pw = pbuf[wave];
#pragma unroll
    for (int r = 0; r < 8; ++r) {
      float s0 = sacc[0][r] * scale;
      float s1 = sacc[1][r] * scale;
      float bm = rowmax16(fmaxf(s0, s1));
      float mn = fmaxf(mrow[r], bm);
      float corr = __expf(mrow[r] - mn);
      float p0 = __expf(s0 - mn);
      float p1 = __expf(s1 - mn);
      float bs = rowsum16(p0 + p1);
      lrow[r] = lrow[r] * corr + bs;
      mrow[r] = mn;
#pragma unroll
      for (int t = 0; t < 4; ++t) o[t][r] *= corr;
      int m = r + 8 * half;
      pw[m * 32 + nl]      = f2bf(p0);
      pw[m * 32 + 16 + nl] = f2bf(p1);
    }
    wait_ds0();                         // intra-wave P visibility (DS in-order)
    v16bf aP = load_a_bf16(pw, 32, 0, 0);
#pragma unroll
    for (int t = 0; t < 4; ++t) {
      v16bf bv = load_b_bf16(vbuf[cur], 32, 16 * t, 0);
      o[t] = WMMA_BF16(aP, bv, o[t]);
    }

    if (s + 1 < NS) wait_async0();
    __syncthreads();
  }

  // normalize, write context [b][seq][DIM] bf16 (DIM index = h*64 + d)
#pragma unroll
  for (int t = 0; t < 4; ++t) {
    int col = h * HD + 16 * t + nl;
#pragma unroll
    for (int r = 0; r < 8; ++r) {
      int m = row0 + r + 8 * half;
      float val = o[t][r] / lrow[r];
      Xctx[((size_t)batch * SEQ + m) * DIM + col] = f2bf(val);
    }
  }
}

// ---------------------------------------------------------------------------
// Kernel 4: output projection  out = Xctx * Wo^T + bo (fp32 out), same tiling.
__global__ void ca_outproj(const __bf16* __restrict__ X, const __bf16* __restrict__ W,
                           const float* __restrict__ bias, float* __restrict__ out) {
  __shared__ __align__(16) __bf16 bbuf[2][64 * 32];

  int tid  = threadIdx.x;
  int wave = tid >> 5;
  int lane = tid & 31;
  int half = lane >> 4, nl = lane & 15;
  int row0 = blockIdx.x * 128 + wave * 16;
  int col0 = blockIdx.y * 64;

  const __bf16* wsrc = W + (size_t)(col0 + (tid >> 2)) * DIM + (tid & 3) * 8;
  __bf16* ldst0 = &bbuf[0][tid * 8];
  __bf16* ldst1 = &bbuf[1][tid * 8];

  v8f zero = {0.f, 0.f, 0.f, 0.f, 0.f, 0.f, 0.f, 0.f};
  v8f acc[4];
#pragma unroll
  for (int t = 0; t < 4; ++t) acc[t] = zero;

  async_cp16(wsrc, ldst0);
  wait_async0();
  __syncthreads();

  const int NK = DIM / 32;
  for (int ks = 0; ks < NK; ++ks) {
    int k = ks * 32;
    const __bf16* bcur = bbuf[ks & 1];
    if (ks + 1 < NK)
      async_cp16(wsrc + k + 32, (ks & 1) ? ldst0 : ldst1);

    v16bf a = load_a_bf16(X, DIM, row0, k);
#pragma unroll
    for (int t = 0; t < 4; ++t) {
      v16bf b = load_b_bf16(bcur, 32, 16 * t, 0);
      acc[t] = WMMA_BF16(a, b, acc[t]);
    }
    if (ks + 1 < NK) wait_async0();
    __syncthreads();
  }

#pragma unroll
  for (int t = 0; t < 4; ++t) {
    int col = col0 + 16 * t + nl;
    float bv = bias[col];
#pragma unroll
    for (int r = 0; r < 8; ++r) {
      int m = row0 + r + 8 * half;
      out[(size_t)m * DIM + col] = acc[t][r] + bv;
    }
  }
}

// ---------------------------------------------------------------------------
extern "C" void kernel_launch(void* const* d_in, const int* in_sizes, int n_in,
                              void* d_out, int out_size, void* d_ws, size_t ws_size,
                              hipStream_t stream) {
  const float* q  = (const float*)d_in[0];
  const float* k  = (const float*)d_in[1];
  const float* v  = (const float*)d_in[2];
  const float* Wq = (const float*)d_in[3];
  const float* bq = (const float*)d_in[4];
  const float* Wk = (const float*)d_in[5];
  const float* bk = (const float*)d_in[6];
  const float* Wv = (const float*)d_in[7];
  const float* bv = (const float*)d_in[8];
  const float* Wo = (const float*)d_in[9];
  const float* bo = (const float*)d_in[10];
  float* out = (float*)d_out;

  const size_t SZ_ACT = (size_t)MROWS * DIM * sizeof(__bf16);  // 16 MiB
  const size_t SZ_W   = (size_t)DIM * DIM * sizeof(__bf16);    //  2 MiB
  char* w = (char*)d_ws;
  __bf16* bfq  = (__bf16*)w; w += SZ_ACT;
  __bf16* bfk  = (__bf16*)w; w += SZ_ACT;
  __bf16* bfv  = (__bf16*)w; w += SZ_ACT;
  __bf16* bWq  = (__bf16*)w; w += SZ_W;
  __bf16* bWk  = (__bf16*)w; w += SZ_W;
  __bf16* bWv  = (__bf16*)w; w += SZ_W;
  __bf16* bWo  = (__bf16*)w; w += SZ_W;
  __bf16* Qh   = (__bf16*)w; w += SZ_ACT;   // [b][h][seq][64]
  __bf16* Kh   = (__bf16*)w; w += SZ_ACT;   // [b][h][seq][64]
  __bf16* Vt   = (__bf16*)w; w += SZ_ACT;   // [b][h][64][seq]
  __bf16* Xctx = (__bf16*)w; w += SZ_ACT;   // [b][seq][1024]

  const int nAct = MROWS * DIM;
  const int nW   = DIM * DIM;
  ca_cvt_bf16<<<(nAct + 255) / 256, 256, 0, stream>>>(q,  bfq, nAct);
  ca_cvt_bf16<<<(nAct + 255) / 256, 256, 0, stream>>>(k,  bfk, nAct);
  ca_cvt_bf16<<<(nAct + 255) / 256, 256, 0, stream>>>(v,  bfv, nAct);
  ca_cvt_bf16<<<(nW   + 255) / 256, 256, 0, stream>>>(Wq, bWq, nW);
  ca_cvt_bf16<<<(nW   + 255) / 256, 256, 0, stream>>>(Wk, bWk, nW);
  ca_cvt_bf16<<<(nW   + 255) / 256, 256, 0, stream>>>(Wv, bWv, nW);
  ca_cvt_bf16<<<(nW   + 255) / 256, 256, 0, stream>>>(Wo, bWo, nW);

  dim3 gProj(MROWS / 128, DIM / 64);         // 64 x 16, 256-thread blocks
  ca_proj<<<gProj, 256, 0, stream>>>(bfq, bWq, bq, Qh, 0);
  ca_proj<<<gProj, 256, 0, stream>>>(bfk, bWk, bk, Kh, 0);
  ca_proj<<<gProj, 256, 0, stream>>>(bfv, bWv, bv, Vt, 1);

  dim3 gAttn(SEQ / 128, BATCH * HEADS);      // 16 x 64, 256-thread blocks
  ca_attn<<<gAttn, 256, 0, stream>>>(Qh, Kh, Vt, Xctx);

  ca_outproj<<<gProj, 256, 0, stream>>>(Xctx, bWo, bo, out);
}
